// DebertaLayer_78726750536333
// MI455X (gfx1250) — matrix operand order repaired and checked by reading, not compiled, not run on hardware
//
#include <hip/hip_runtime.h>
#include <math.h>

// ---------------- problem constants ----------------
#define Bn   16
#define Sn   512
#define Hn   768
#define NHn  12
#define DHn  64
#define Pn   512          // relative position clip (identity for S=512)
#define TWOP 1024
#define In   3072
#define BS   (Bn*Sn)      // 8192 rows
#define ATT_SCALE 0.07216878364870322f   // 1/sqrt(3*64)

typedef __bf16 bf16;
typedef __attribute__((ext_vector_type(16))) __bf16 v16bf;
typedef __attribute__((ext_vector_type(8)))  __bf16 bf16x8;
typedef __attribute__((ext_vector_type(4)))  __bf16 bf16x4;
typedef __attribute__((ext_vector_type(8)))  float  v8f;
typedef __attribute__((ext_vector_type(4)))  float  f32x4;

union ABFrag { v16bf v; bf16x8 h[2]; };

__device__ inline v8f wmma_bf16(const ABFrag& a, const ABFrag& b, v8f c) {
  return __builtin_amdgcn_wmma_f32_16x16x32_bf16(
      /*neg_a=*/false, a.v, /*neg_b=*/false, b.v,
      /*c_mod=*/(short)0, c, /*reuse_a=*/false, /*reuse_b=*/false);
}

// async 16B/lane copy global -> LDS (ASYNCcnt-tracked, no VGPR data path)
__device__ inline void async_copy_b128(void* lds_ptr, const void* gptr) {
  uint32_t lo = (uint32_t)(uintptr_t)lds_ptr;   // low 32 bits of flat = LDS offset
  asm volatile("global_load_async_to_lds_b128 %0, %1, off"
               :: "v"(lo), "v"(gptr) : "memory");
}

// ---------------- fp32 -> bf16 convert (vectorized x4) ----------------
__global__ __launch_bounds__(256)
void cvt_bf16_kernel(const float* __restrict__ src, bf16* __restrict__ dst, int n4) {
  int i = blockIdx.x * 256 + threadIdx.x;
  if (i < n4) {
    f32x4 f = *(const f32x4*)(src + (size_t)i * 4);
    bf16x4 o;
    o.x = (bf16)f.x; o.y = (bf16)f.y; o.z = (bf16)f.z; o.w = (bf16)f.w;
    *(bf16x4*)(dst + (size_t)i * 4) = o;
  }
}

// ---------------- transpose + convert: Wt[n*K+k] = bf16(W[k*N+n]) ----------------
__global__ __launch_bounds__(256)
void transpose_cvt_kernel(const float* __restrict__ W, bf16* __restrict__ Wt,
                          int K, int N) {
  long i = (long)blockIdx.x * 256 + threadIdx.x;
  long tot = (long)K * N;
  if (i < tot) {
    int n = (int)(i / K);
    int k = (int)(i % K);
    Wt[i] = (bf16)W[(size_t)k * N + n];
  }
}

// ---------------- generic bf16 WMMA GEMM ----------------
// C[M,N] = act(A[M,K] * Bt[N,K]^T + bias).  64x128 block tile, 8 waves (4Mx2N),
// K step 32, double-buffered async global->LDS staging, fp32 accumulate.
template<int WRITE_F32, int WRITE_BF16, int DO_GELU>
__global__ __launch_bounds__(256)
void gemm_bf16_kernel(const bf16* __restrict__ A, const bf16* __restrict__ Bt,
                      const float* __restrict__ bias,
                      float* __restrict__ Cf, bf16* __restrict__ Cb,
                      int M, int N, int K) {
  __shared__ bf16 sA[2][64 * 40];    // 64 rows  x 32 K (+8 pad)
  __shared__ bf16 sB[2][128 * 40];   // 128 N-rows x 32 K (+8 pad)

  const int tid  = threadIdx.x;
  const int m0   = blockIdx.x * 64;
  const int n0   = blockIdx.y * 128;
  const int wv   = tid >> 5;
  const int lane = tid & 31;
  const int wm   = wv & 3;       // M tile 0..3
  const int wn   = wv >> 2;      // N half 0..1 (64 cols each)
  const int lrow = lane & 15;
  const int sel  = lane >> 4;

  const int ar = tid >> 2;       // A coop row 0..63
  const int ac = (tid & 3) * 8;  // A coop K offset

  v8f acc[4] = {{}, {}, {}, {}};

  auto stage = [&](int buf, int k0) {
    async_copy_b128(&sA[buf][ar * 40 + ac],
                    A + (size_t)(m0 + ar) * K + k0 + ac);
#pragma unroll
    for (int c = 0; c < 2; ++c) {
      const int id = tid + c * 256;       // 0..511
      const int r  = id >> 2;             // B row 0..127
      const int cc = (id & 3) * 8;
      async_copy_b128(&sB[buf][r * 40 + cc],
                      Bt + (size_t)(n0 + r) * K + k0 + cc);
    }
  };

  stage(0, 0);
  int buf = 0;
  for (int k0 = 0; k0 < K; k0 += 32, buf ^= 1) {
    if (k0 + 32 < K) {
      stage(buf ^ 1, k0 + 32);
      asm volatile("s_wait_asynccnt 0x3" ::: "memory");  // current stage's 3 done
    } else {
      asm volatile("s_wait_asynccnt 0x0" ::: "memory");
    }
    __syncthreads();

    ABFrag a;
    a.h[0] = *(const bf16x8*)(&sA[buf][(wm * 16 + lrow) * 40 + sel * 16]);
    a.h[1] = *(const bf16x8*)(&sA[buf][(wm * 16 + lrow) * 40 + sel * 16 + 8]);
#pragma unroll
    for (int j = 0; j < 4; ++j) {
      const int nb = wn * 64 + j * 16 + lrow;
      ABFrag bfr;
      bfr.h[0] = *(const bf16x8*)(&sB[buf][nb * 40 + sel * 16]);
      bfr.h[1] = *(const bf16x8*)(&sB[buf][nb * 40 + sel * 16 + 8]);
      acc[j] = wmma_bf16(a, bfr, acc[j]);
    }
    __syncthreads();
  }

#pragma unroll
  for (int j = 0; j < 4; ++j) {
    const int col = n0 + wn * 64 + j * 16 + lrow;
#pragma unroll
    for (int r = 0; r < 8; ++r) {
      const int row = m0 + wm * 16 + r + 8 * sel;
      float v = acc[j][r] + (bias ? bias[col] : 0.f);
      if (DO_GELU) v = 0.5f * v * (1.f + erff(v * 0.7071067811865476f));
      if (WRITE_F32)  Cf[(size_t)row * N + col] = v;
      if (WRITE_BF16) Cb[(size_t)row * N + col] = (bf16)v;
    }
  }
}

// ---------------- fused disentangled attention ----------------
// block = (i-block of 64 query rows, head h, batch b); 256 threads = 8 waves.
// LDS: scores 64x512 f32, probs 64x512 bf16, per-wave G scratch (2 x 16x32 f32).
#define SSTRIDE 516
#define PSTRIDE 520
#define GSTRIDE 36
#define SS_BYTES (64 * SSTRIDE * 4)            // 132096
#define SP_BYTES (64 * PSTRIDE * 2)            // 66560
#define SG_BYTES (8 * 2 * 16 * GSTRIDE * 4)    // 36864
#define ATTN_LDS (SS_BYTES + SP_BYTES + SG_BYTES)

__global__ __launch_bounds__(256)
void attn_kernel(const bf16* __restrict__ qb, const bf16* __restrict__ kb,
                 const bf16* __restrict__ vb,
                 const bf16* __restrict__ poskb, const bf16* __restrict__ posqb,
                 const float* __restrict__ mask, bf16* __restrict__ ctxb) {
  extern __shared__ char smem[];
  float* sS = (float*)smem;
  bf16*  sP = (bf16*)(smem + SS_BYTES);
  float* sG = (float*)(smem + SS_BYTES + SP_BYTES);

  const int tid  = threadIdx.x;
  const int wv   = tid >> 5;
  const int lane = tid & 31;
  const int lrow = lane & 15;
  const int sel  = lane >> 4;

  const int i0b = blockIdx.x * 64;
  const int h   = blockIdx.y;
  const int b   = blockIdx.z;
  const int bS  = b * Sn;
  const int hoff = h * DHn;

  // ---- stage 1: scores = c2c + c2p + p2c (scaled, masked) ----
  const int mt    = wv & 3;          // which 16-row query tile of the 64
  const int jhalf = wv >> 2;         // split 32 j-tiles between wave pairs
  const int i0t   = i0b + mt * 16;

  // Q tile A-fragments (invariant over j)
  ABFrag qa[2];
  {
    const bf16* qp = qb + (size_t)(bS + i0t + lrow) * Hn + hoff;
#pragma unroll
    for (int ks = 0; ks < 2; ++ks) {
      qa[ks].h[0] = *(const bf16x8*)(qp + ks * 32 + sel * 16);
      qa[ks].h[1] = *(const bf16x8*)(qp + ks * 32 + sel * 16 + 8);
    }
  }

  float* Gc = sG + wv * (2 * 16 * GSTRIDE);
  float* Gp = Gc + 16 * GSTRIDE;

  for (int jt = jhalf; jt < 32; jt += 2) {
    const int j0 = jt * 16;
    const bf16* kp = kb + (size_t)(bS + j0 + lrow) * Hn + hoff;

    ABFrag kbf[2], kaf[2];
#pragma unroll
    for (int ks = 0; ks < 2; ++ks) {
      // B-layout fragment of K^T (lane = key row, contiguous 16 of DH)
      kbf[ks].h[0] = *(const bf16x8*)(kp + ks * 32 + sel * 16);
      kbf[ks].h[1] = *(const bf16x8*)(kp + ks * 32 + sel * 16 + 8);
      // A-layout fragment of K tile (for p2c GEMM)
      kaf[ks].h[0] = *(const bf16x8*)(kp + ks * 32 + sel * 16);
      kaf[ks].h[1] = *(const bf16x8*)(kp + ks * 32 + sel * 16 + 8);
    }

    v8f cc = {};
#pragma unroll
    for (int ks = 0; ks < 2; ++ks) cc = wmma_bf16(qa[ks], kbf[ks], cc);

    // c2p:  Gc[u,t] = q[i0t+u] . pos_k[base1-15+t],  t in [0,32)
    // p2c:  Gp[w,t] = k[j0 +w] . pos_q[base2-15+t]
    const int base1 = i0t - j0 + Pn;
    const int base2 = j0 - i0t + Pn;
    v8f gc[2] = {{}, {}};
    v8f gp[2] = {{}, {}};
#pragma unroll
    for (int nt = 0; nt < 2; ++nt) {
      int pr1 = base1 - 15 + nt * 16 + lrow; pr1 = pr1 < 0 ? 0 : (pr1 > TWOP - 1 ? TWOP - 1 : pr1);
      int pr2 = base2 - 15 + nt * 16 + lrow; pr2 = pr2 < 0 ? 0 : (pr2 > TWOP - 1 ? TWOP - 1 : pr2);
      const bf16* pk = poskb + (size_t)pr1 * Hn + hoff;
      const bf16* pq = posqb + (size_t)pr2 * Hn + hoff;
#pragma unroll
      for (int ks = 0; ks < 2; ++ks) {
        ABFrag pb, qfb;
        pb.h[0]  = *(const bf16x8*)(pk + ks * 32 + sel * 16);
        pb.h[1]  = *(const bf16x8*)(pk + ks * 32 + sel * 16 + 8);
        qfb.h[0] = *(const bf16x8*)(pq + ks * 32 + sel * 16);
        qfb.h[1] = *(const bf16x8*)(pq + ks * 32 + sel * 16 + 8);
        gc[nt] = wmma_bf16(qa[ks],  pb,  gc[nt]);
        gp[nt] = wmma_bf16(kaf[ks], qfb, gp[nt]);
      }
    }

    // dump G matrices to this wave's LDS scratch, then pick diagonals
#pragma unroll
    for (int nt = 0; nt < 2; ++nt)
#pragma unroll
      for (int r = 0; r < 8; ++r) {
        Gc[(r + 8 * sel) * GSTRIDE + nt * 16 + lrow] = gc[nt][r];
        Gp[(r + 8 * sel) * GSTRIDE + nt * 16 + lrow] = gp[nt][r];
      }
    asm volatile("s_wait_dscnt 0x0" ::: "memory");

    const int w = lrow;
    const float mval = mask[bS + j0 + w];
#pragma unroll
    for (int r = 0; r < 8; ++r) {
      const int u = r + 8 * sel;
      float val = cc[r] + Gc[u * GSTRIDE + (u + 15 - w)] + Gp[w * GSTRIDE + (w + 15 - u)];
      val = (mval > 0.f) ? val * ATT_SCALE : -1e9f;
      sS[(mt * 16 + u) * SSTRIDE + j0 + w] = val;
    }
  }
  __syncthreads();

  // ---- stage 2: row softmax (4 threads per row, quad shuffles) ----
  {
    const int row = tid >> 2;
    const int qd  = tid & 3;
    float* rp = sS + row * SSTRIDE + qd * 128;
    float mx = -3.4e38f;
    for (int c = 0; c < 128; ++c) mx = fmaxf(mx, rp[c]);
    mx = fmaxf(mx, __shfl_xor(mx, 1, 32));
    mx = fmaxf(mx, __shfl_xor(mx, 2, 32));
    float sum = 0.f;
    for (int c = 0; c < 128; ++c) { float e = expf(rp[c] - mx); rp[c] = e; sum += e; }
    sum += __shfl_xor(sum, 1, 32);
    sum += __shfl_xor(sum, 2, 32);
    const float inv = 1.f / sum;
    bf16* pp = sP + row * PSTRIDE + qd * 128;
    for (int c = 0; c < 128; ++c) pp[c] = (bf16)(rp[c] * inv);
  }
  __syncthreads();

  // ---- stage 3: ctx = probs(64x512) x V(512x64) ----
  for (int t = wv; t < 16; t += 8) {
    const int mtt = t & 3;
    const int nt  = t >> 2;
    v8f acc = {};
    const bf16* ap = sP + (mtt * 16 + lrow) * PSTRIDE;
    const int d = hoff + nt * 16 + lrow;
    for (int k0 = 0; k0 < Sn; k0 += 32) {
      ABFrag a, bfr;
      a.h[0] = *(const bf16x8*)(ap + k0 + sel * 16);
      a.h[1] = *(const bf16x8*)(ap + k0 + sel * 16 + 8);
#pragma unroll
      for (int e = 0; e < 16; ++e) {
        const int kk = k0 + sel * 16 + e;
        ((bf16*)&bfr)[e] = vb[(size_t)(bS + kk) * Hn + d];
      }
      acc = wmma_bf16(a, bfr, acc);
    }
#pragma unroll
    for (int r = 0; r < 8; ++r) {
      const int row = i0b + mtt * 16 + r + 8 * sel;
      ctxb[(size_t)(bS + row) * Hn + d] = (bf16)acc[r];
    }
  }
}

// ---------------- residual add + LayerNorm (H=768, 1 block/row) ----------------
__global__ __launch_bounds__(256)
void add_ln_kernel(const float* __restrict__ x, const float* __restrict__ y,
                   const float* __restrict__ g, const float* __restrict__ bb,
                   float* __restrict__ outf, bf16* __restrict__ outb) {
  const int row = blockIdx.x;
  const int tid = threadIdx.x;
  __shared__ float red[8], red2[8];

  float v[3]; float s = 0.f;
#pragma unroll
  for (int i = 0; i < 3; ++i) {
    const int c = tid + i * 256;
    v[i] = x[(size_t)row * Hn + c] + y[(size_t)row * Hn + c];
    s += v[i];
  }
  for (int o = 16; o; o >>= 1) s += __shfl_xor(s, o, 32);
  if ((tid & 31) == 0) red[tid >> 5] = s;
  __syncthreads();
  float mean = 0.f;
#pragma unroll
  for (int i = 0; i < 8; ++i) mean += red[i];
  mean *= (1.f / Hn);

  float sq = 0.f;
#pragma unroll
  for (int i = 0; i < 3; ++i) { const float d = v[i] - mean; sq += d * d; }
  for (int o = 16; o; o >>= 1) sq += __shfl_xor(sq, o, 32);
  if ((tid & 31) == 0) red2[tid >> 5] = sq;
  __syncthreads();
  float var = 0.f;
#pragma unroll
  for (int i = 0; i < 8; ++i) var += red2[i];
  var *= (1.f / Hn);
  const float r = rsqrtf(var + 1e-7f);

#pragma unroll
  for (int i = 0; i < 3; ++i) {
    const int c = tid + i * 256;
    const float o = (v[i] - mean) * r * g[c] + bb[c];
    outf[(size_t)row * Hn + c] = o;
    if (outb) outb[(size_t)row * Hn + c] = (bf16)o;
  }
}

// ---------------- host orchestration ----------------
extern "C" void kernel_launch(void* const* d_in, const int* in_sizes, int n_in,
                              void* d_out, int out_size, void* d_ws, size_t ws_size,
                              hipStream_t stream) {
  (void)in_sizes; (void)n_in; (void)out_size; (void)ws_size;
  const float* hs   = (const float*)d_in[0];
  const float* mask = (const float*)d_in[1];
  const float* pos  = (const float*)d_in[2];
  const float* Wq = (const float*)d_in[3];  const float* bq = (const float*)d_in[4];
  const float* Wk = (const float*)d_in[5];  const float* bk = (const float*)d_in[6];
  const float* Wv = (const float*)d_in[7];  const float* bv = (const float*)d_in[8];
  const float* Wpk = (const float*)d_in[9];
  const float* Wpq = (const float*)d_in[10];
  const float* Wo = (const float*)d_in[11]; const float* bo = (const float*)d_in[12];
  const float* g1 = (const float*)d_in[13]; const float* be1 = (const float*)d_in[14];
  const float* W1 = (const float*)d_in[15]; const float* b1 = (const float*)d_in[16];
  const float* W2 = (const float*)d_in[17]; const float* b2 = (const float*)d_in[18];
  const float* g2 = (const float*)d_in[19]; const float* be2 = (const float*)d_in[20];
  float* out = (float*)d_out;

  char* ws = (char*)d_ws; size_t off = 0;
  auto alloc = [&](size_t bytes) -> char* {
    char* p = ws + off; off += (bytes + 255) & ~(size_t)255; return p;
  };

  bf16* hs_bf   = (bf16*)alloc((size_t)BS * Hn * 2);
  bf16* pos_bf  = (bf16*)alloc((size_t)TWOP * Hn * 2);
  bf16* Wq_t    = (bf16*)alloc((size_t)Hn * Hn * 2);
  bf16* Wk_t    = (bf16*)alloc((size_t)Hn * Hn * 2);
  bf16* Wv_t    = (bf16*)alloc((size_t)Hn * Hn * 2);
  bf16* Wpk_t   = (bf16*)alloc((size_t)Hn * Hn * 2);
  bf16* Wpq_t   = (bf16*)alloc((size_t)Hn * Hn * 2);
  bf16* Wo_t    = (bf16*)alloc((size_t)Hn * Hn * 2);
  bf16* W1_t    = (bf16*)alloc((size_t)In * Hn * 2);   // [N=I][K=H]
  bf16* W2_t    = (bf16*)alloc((size_t)Hn * In * 2);   // [N=H][K=I]
  bf16* q_bf    = (bf16*)alloc((size_t)BS * Hn * 2);
  bf16* k_bf    = (bf16*)alloc((size_t)BS * Hn * 2);
  bf16* v_bf    = (bf16*)alloc((size_t)BS * Hn * 2);
  bf16* posk_bf = (bf16*)alloc((size_t)TWOP * Hn * 2);
  bf16* posq_bf = (bf16*)alloc((size_t)TWOP * Hn * 2);
  bf16* ctx_bf  = (bf16*)alloc((size_t)BS * Hn * 2);
  float* attn_f = (float*)alloc((size_t)BS * Hn * 4);
  float* h1_f   = (float*)alloc((size_t)BS * Hn * 4);
  bf16*  h1_bf  = (bf16*)alloc((size_t)BS * Hn * 2);
  bf16*  mid_bf = (bf16*)alloc((size_t)BS * In * 2);
  float* ffn_f  = (float*)alloc((size_t)BS * Hn * 4);

  // 1. convert activations / positions
  {
    int n4 = BS * Hn / 4;
    cvt_bf16_kernel<<<(n4 + 255) / 256, 256, 0, stream>>>(hs, hs_bf, n4);
    n4 = TWOP * Hn / 4;
    cvt_bf16_kernel<<<(n4 + 255) / 256, 256, 0, stream>>>(pos, pos_bf, n4);
  }
  // 2. transpose+convert weights to [N][K]
  {
    const long hh = (long)Hn * Hn; const int gb = (int)((hh + 255) / 256);
    transpose_cvt_kernel<<<gb, 256, 0, stream>>>(Wq,  Wq_t,  Hn, Hn);
    transpose_cvt_kernel<<<gb, 256, 0, stream>>>(Wk,  Wk_t,  Hn, Hn);
    transpose_cvt_kernel<<<gb, 256, 0, stream>>>(Wv,  Wv_t,  Hn, Hn);
    transpose_cvt_kernel<<<gb, 256, 0, stream>>>(Wpk, Wpk_t, Hn, Hn);
    transpose_cvt_kernel<<<gb, 256, 0, stream>>>(Wpq, Wpq_t, Hn, Hn);
    transpose_cvt_kernel<<<gb, 256, 0, stream>>>(Wo,  Wo_t,  Hn, Hn);
    const long hi = (long)Hn * In; const int gi = (int)((hi + 255) / 256);
    transpose_cvt_kernel<<<gi, 256, 0, stream>>>(W1, W1_t, Hn, In);  // K=H,N=I
    transpose_cvt_kernel<<<gi, 256, 0, stream>>>(W2, W2_t, In, Hn);  // K=I,N=H
  }
  // 3. QKV + positional projections (bf16 outputs)
  {
    dim3 g(BS / 64, Hn / 128);
    gemm_bf16_kernel<0,1,0><<<g, 256, 0, stream>>>(hs_bf, Wq_t, bq, nullptr, q_bf, BS, Hn, Hn);
    gemm_bf16_kernel<0,1,0><<<g, 256, 0, stream>>>(hs_bf, Wk_t, bk, nullptr, k_bf, BS, Hn, Hn);
    gemm_bf16_kernel<0,1,0><<<g, 256, 0, stream>>>(hs_bf, Wv_t, bv, nullptr, v_bf, BS, Hn, Hn);
    dim3 gp(TWOP / 64, Hn / 128);
    gemm_bf16_kernel<0,1,0><<<gp, 256, 0, stream>>>(pos_bf, Wpk_t, nullptr, nullptr, posk_bf, TWOP, Hn, Hn);
    gemm_bf16_kernel<0,1,0><<<gp, 256, 0, stream>>>(pos_bf, Wpq_t, nullptr, nullptr, posq_bf, TWOP, Hn, Hn);
  }
  // 4. fused disentangled attention -> ctx (bf16)
  attn_kernel<<<dim3(Sn / 64, NHn, Bn), 256, ATTN_LDS, stream>>>(
      q_bf, k_bf, v_bf, posk_bf, posq_bf, mask, ctx_bf);
  // 5. output projection (fp32), residual + LN -> h1
  gemm_bf16_kernel<1,0,0><<<dim3(BS / 64, Hn / 128), 256, 0, stream>>>(
      ctx_bf, Wo_t, bo, attn_f, nullptr, BS, Hn, Hn);
  add_ln_kernel<<<BS, 256, 0, stream>>>(hs, attn_f, g1, be1, h1_f, h1_bf);
  // 6. FFN: GELU(h1 W1 + b1) W2 + b2, residual + LN -> out
  gemm_bf16_kernel<0,1,1><<<dim3(BS / 64, In / 128), 256, 0, stream>>>(
      h1_bf, W1_t, b1, nullptr, mid_bf, BS, In, Hn);
  gemm_bf16_kernel<1,0,0><<<dim3(BS / 64, Hn / 128), 256, 0, stream>>>(
      mid_bf, W2_t, b2, ffn_f, nullptr, BS, Hn, In);
  add_ln_kernel<<<BS, 256, 0, stream>>>(h1_f, ffn_f, g2, be2, out, nullptr);
}
